// SillyDNA2RNA_23682449670388
// MI455X (gfx1250) — compile-verified, matrix operand order checked by reading, not run on hardware
//
#include <hip/hip_runtime.h>

// ---------------------------------------------------------------------------
// Per-head MLP: h = relu(h @ W + b) chain, heads O=2000, batch B=128.
// Memory-bound on the 280MB weight stream -> stage weights once into LDS per
// head, do all GEMMs with v_wmma_f32_16x16x32_f16 (f16 in, f32 accumulate).
// ---------------------------------------------------------------------------

typedef _Float16 half_t;
typedef __attribute__((ext_vector_type(16))) _Float16 v16h;
typedef __attribute__((ext_vector_type(8)))  float    v8f;

#define BATCH    128
#define NHEADS   2000
#define LGATHER  64
#define INDIM    100000
#define ASTRIDE  136   // halves; multiple of 8 (16B frag alignment), odd*8 -> no bank conflicts
#define WSTRIDE  136

// layer dims: 64 -> 91 -> 128 -> 91 -> 64 -> 1, padded to 16/32 multiples
constexpr int DIN [5] = { 64,  91, 128,  91, 64 };
constexpr int DOUT[5] = { 91, 128,  91,  64,  1 };
constexpr int PIN [5] = { 64,  96, 128,  96, 64 };   // K padded (mult of 32)
constexpr int POUT[5] = { 96, 128,  96,  64, 16 };   // N padded (mult of 16)

struct MlpParams {
  const float*  x;       // (128, 100000)
  const half_t* xT;      // (100000, 128) f16, or nullptr
  const int*    idxs;    // (2000, 64)
  const float*  W[5];    // (2000, d_i, d_{i+1})
  const float*  bias[5]; // (2000, d_{i+1})
  float*        out;     // (128, 2000)
};

// --- prep: transpose + f32->f16 convert x so the gather is coalesced --------
__global__ __launch_bounds__(256) void dna_transpose_x(const float* __restrict__ x,
                                                       half_t* __restrict__ xT) {
  __shared__ float tile[32][33];
  const int i0 = blockIdx.x * 32;
  const int b0 = blockIdx.y * 32;
  const int li = threadIdx.x & 31;
  const int ls = threadIdx.x >> 5;   // 8 rows per pass
  for (int bb = ls; bb < 32; bb += 8) {
    int i = i0 + li, b = b0 + bb;
    tile[bb][li] = (i < INDIM) ? x[(size_t)b * INDIM + i] : 0.0f;
  }
  __syncthreads();
  for (int ii = ls; ii < 32; ii += 8) {
    int i = i0 + ii, b = b0 + li;
    if (i < INDIM) xT[(size_t)i * BATCH + b] = (half_t)tile[li][ii];
  }
}

// --- main: one workgroup (8 wave32) per head --------------------------------
__global__ __launch_bounds__(256) void dna_mlp_heads(MlpParams p) {
  __shared__ half_t act[2][BATCH * ASTRIDE];   // ping-pong activations (f16)
  __shared__ half_t wt[128 * WSTRIDE];         // weights, transposed [n][k] (f16)
  __shared__ float  bsh[128];                  // bias, zero-padded

  const int o    = blockIdx.x;
  const int tid  = threadIdx.x;
  const int lane = tid & 31;
  const int wave = tid >> 5;
  const int g    = lane >> 4;   // half-wave group
  const int mn   = lane & 15;   // M for A-frag, N for B-frag / C-frag

  // Stage 0: gather x[:, idxs[o]] -> act[0] (B x 64, f16)
  const int* idx_o = p.idxs + o * LGATHER;
  if (p.xT) {
    for (int t = tid; t < BATCH * LGATHER; t += 256) {
      int b = t & (BATCH - 1), l = t >> 7;
      act[0][b * ASTRIDE + l] = p.xT[(size_t)idx_o[l] * BATCH + b];  // coalesced in b
    }
  } else {
    for (int t = tid; t < BATCH * LGATHER; t += 256) {
      int b = t & (BATCH - 1), l = t >> 7;
      act[0][b * ASTRIDE + l] = (half_t)p.x[(size_t)b * INDIM + idx_o[l]];
    }
  }
  __syncthreads();

#pragma unroll
  for (int layer = 0; layer < 5; ++layer) {
    const int DI = DIN[layer], DO = DOUT[layer];
    const int PI = PIN[layer], PO = POUT[layer];
    const int ia = layer & 1, ib = ia ^ 1;
    const float* Wg = p.W[layer]    + (size_t)o * DI * DO;
    const float* bg = p.bias[layer] + (size_t)o * DO;

    // Stage weights transposed+padded into LDS: wt[n*WSTRIDE + k]
    for (int t = tid; t < PI * PO; t += 256) {      // counts divisible by 256
      int c = t % PO, r = t / PO;                   // c fastest -> coalesced global
      float w = (r < DI && c < DO) ? Wg[r * DO + c] : 0.0f;
      wt[c * WSTRIDE + r] = (half_t)w;
    }
    for (int c = tid; c < PO; c += 256) bsh[c] = (c < DO) ? bg[c] : 0.0f;
    __syncthreads();

    const int mbase  = wave * 16;                   // each wave owns a 16-row batch tile
    const int ntiles = PO >> 4;
    const int ktiles = PI >> 5;
    for (int nt = 0; nt < ntiles; ++nt) {
      v8f acc = {};
      for (int kt = 0; kt < ktiles; ++kt) {
        // 16-bit operand layout: lane holds row mn, K-halves {g*8..g*8+7} and +16
        union { v16h v; uint4 q[2]; } af, bf;
        const int kb = kt * 32 + g * 8;
        af.q[0] = *(const uint4*)&act[ia][(mbase + mn) * ASTRIDE + kb];
        af.q[1] = *(const uint4*)&act[ia][(mbase + mn) * ASTRIDE + kb + 16];
        bf.q[0] = *(const uint4*)&wt[(nt * 16 + mn) * WSTRIDE + kb];
        bf.q[1] = *(const uint4*)&wt[(nt * 16 + mn) * WSTRIDE + kb + 16];
        acc = __builtin_amdgcn_wmma_f32_16x16x32_f16(
            false, af.v, false, bf.v, (short)0, acc, false, false);
      }
      const float bv = bsh[nt * 16 + mn];
      if (layer < 4) {
#pragma unroll
        for (int v = 0; v < 8; ++v) {               // C layout: row g*8+v, col mn
          float r = fmaxf(acc[v] + bv, 0.0f);
          act[ib][(mbase + g * 8 + v) * ASTRIDE + nt * 16 + mn] = (half_t)r;
        }
      } else if (mn == 0) {                         // final layer: only column 0 is real
#pragma unroll
        for (int v = 0; v < 8; ++v)
          p.out[(size_t)(mbase + g * 8 + v) * NHEADS + o] = acc[v] + bv;
      }
    }
    __syncthreads();
  }
}

extern "C" void kernel_launch(void* const* d_in, const int* in_sizes, int n_in,
                              void* d_out, int out_size, void* d_ws, size_t ws_size,
                              hipStream_t stream) {
  (void)in_sizes; (void)n_in; (void)out_size;
  MlpParams p;
  p.x    = (const float*)d_in[0];
  p.idxs = (const int*)d_in[1];
  for (int i = 0; i < 5; ++i) {
    p.W[i]    = (const float*)d_in[2 + i];
    p.bias[i] = (const float*)d_in[7 + i];
  }
  p.out = (float*)d_out;
  p.xT  = nullptr;

  const size_t xtBytes = (size_t)INDIM * BATCH * sizeof(half_t);  // 25.6 MB
  if (ws_size >= xtBytes) {
    half_t* xT = (half_t*)d_ws;
    dim3 tg(INDIM / 32, BATCH / 32);  // 100000 % 32 == 0
    dna_transpose_x<<<tg, dim3(256), 0, stream>>>(p.x, xT);
    p.xT = xT;
  }
  dna_mlp_heads<<<dim3(NHEADS), dim3(256), 0, stream>>>(p);
}
